// HGRNFixed16bit_78950088835413
// MI455X (gfx1250) — 8-bit matrix operand order repaired and checked by reading, compiled, not run on hardware
//
#include <hip/hip_runtime.h>
#include <hip/hip_bf16.h>
#include <stdint.h>


typedef __attribute__((ext_vector_type(8))) int v8i;

#define B_  8
#define S_  2048
#define H_  1024
#define M_  (B_ * S_)
#define MT_ (M_ / 16)
#define NT_ (H_ / 16)
#define KC_ (H_ / 64)

#define WOFF_PW   ((size_t)0)
#define WOFF_ALO  ((size_t)4  * 1024 * 1024)
#define WOFF_AHI  (WOFF_ALO + (size_t)M_ * H_)
#define WOFF_FS   (WOFF_AHI + (size_t)M_ * H_)
#define WOFF_IM   (WOFF_FS + (size_t)M_ * H_ * 2)
#define WOFF_G    (WOFF_IM + (size_t)M_ * H_ * 2)

__device__ __forceinline__ int clip16i(int v) {
    return v > 32767 ? 32767 : (v < -32768 ? -32768 : v);
}
__device__ __forceinline__ int fmul16(int a, int b) {
    return clip16i((a * b) >> 8);
}
__device__ __forceinline__ int sig_node(int idx) {
    float xf = (float)(idx * 64 - 32768) * (1.0f / 256.0f);
    float s  = 1.0f / (1.0f + expf(-xf));
    return clip16i((int)rintf(s * 256.0f));
}
__device__ __forceinline__ int sigq(int v) {
    int xs  = v + 32768;
    int idx = xs >> 6; if (idx > 1022) idx = 1022;
    int rem = xs & 63;
    int y0 = sig_node(idx);
    int y1 = sig_node(idx + 1);
    return (short)(y0 + (((y1 - y0) * rem) >> 6));
}

__device__ __forceinline__ int a_klocal(int lane, int v, int b) {
    return ((v & 4) ? 32 : 0) + ((lane & 16) ? 16 : 0) + ((v & 3) << 2) + b;
}
__device__ __forceinline__ int b_klocal(int lane, int v, int b) {
    return ((v & 4) ? 32 : 0) + ((lane & 16) ? 16 : 0) + ((v & 3) << 2) + b;
}

__global__ void __launch_bounds__(256)
pack_w_kernel(const int* __restrict__ w, int8_t* __restrict__ dst) {
    const int kc   = blockIdx.x;
    const int nt   = blockIdx.y;
    const int lane = threadIdx.x >> 3;
    const int v    = threadIdx.x & 7;
    const int n    = nt * 16 + (lane & 15);
    unsigned int word = 0;
#pragma unroll
    for (int b = 0; b < 4; ++b) {
        int k = kc * 64 + b_klocal(lane, v, b);
        int val = w[(size_t)k * H_ + n];
        word |= ((unsigned int)(val & 0xFF)) << (8 * b);
    }
    unsigned int* blk =
        (unsigned int*)(dst + ((size_t)(nt * KC_ + kc) << 10));
    blk[threadIdx.x] = word;
}

__global__ void __launch_bounds__(256)
quant_pack_x_kernel(const float* __restrict__ x,
                    uint8_t* __restrict__ alo, uint8_t* __restrict__ ahi) {
    const int kc   = blockIdx.x;
    const int mt   = blockIdx.y;
    const int lane = threadIdx.x >> 3;
    const int v    = threadIdx.x & 7;
    const int m    = mt * 16 + (lane & 15);
    unsigned int wlo = 0, whi = 0;
#pragma unroll
    for (int b = 0; b < 4; ++b) {
        int k = kc * 64 + a_klocal(lane, v, b);
        float r = rintf(x[(size_t)m * H_ + k] * 256.0f);
        r = fminf(fmaxf(r, -32768.0f), 32767.0f);
        int q = (int)r;
        wlo |= ((unsigned int)(q & 0xFF)) << (8 * b);
        whi |= ((unsigned int)((q >> 8) & 0xFF)) << (8 * b);
    }
    size_t off = (size_t)(mt * KC_ + kc) << 10;
    ((unsigned int*)(alo + off))[threadIdx.x] = wlo;
    ((unsigned int*)(ahi + off))[threadIdx.x] = whi;
}

__global__ void __launch_bounds__(256)
pack_gh_kernel(const short* __restrict__ gh,
               uint8_t* __restrict__ alo, uint8_t* __restrict__ ahi) {
    const int kc   = blockIdx.x;
    const int mt   = blockIdx.y;
    const int lane = threadIdx.x >> 3;
    const int v    = threadIdx.x & 7;
    const int m    = mt * 16 + (lane & 15);
    unsigned int wlo = 0, whi = 0;
#pragma unroll
    for (int b = 0; b < 4; ++b) {
        int k = kc * 64 + a_klocal(lane, v, b);
        int q = gh[(size_t)m * H_ + k];
        wlo |= ((unsigned int)(q & 0xFF)) << (8 * b);
        whi |= ((unsigned int)((q >> 8) & 0xFF)) << (8 * b);
    }
    size_t off = (size_t)(mt * KC_ + kc) << 10;
    ((unsigned int*)(alo + off))[threadIdx.x] = wlo;
    ((unsigned int*)(ahi + off))[threadIdx.x] = whi;
}

__global__ void __launch_bounds__(256)
gemm_ifg_kernel(const uint8_t* __restrict__ Alo, const uint8_t* __restrict__ Ahi,
                const int8_t* __restrict__ Bi, const int8_t* __restrict__ Bf,
                const int8_t* __restrict__ Bg,
                const int* __restrict__ si, const int* __restrict__ sf,
                const int* __restrict__ sg,
                short* __restrict__ FS, short* __restrict__ IM,
                short* __restrict__ G) {
    __shared__ __align__(16) uint8_t sm[2][14 * 1024];
    const int tid  = threadIdx.x;
    const int lane = tid & 31;
    const int w    = tid >> 5;
    const int mtBase = (blockIdx.x >> 5) << 2;
    const int ntBase = (blockIdx.x & 31) << 1;
    const int mi = w & 3;
    const int ni = w >> 2;

    const int jA   = tid >> 6;
    const int offA = (tid & 63) << 4;
    const int nnB  = tid >> 7;
    const int offB = (tid & 127) << 3;

    auto stage = [&](uint8_t* dst, int kc) {
        const size_t aoff = (((size_t)(mtBase + jA) * KC_ + kc) << 10) + offA;
        const size_t boff = (((size_t)(ntBase + nnB) * KC_ + kc) << 10) + offB;
        const int4 tLo = *(const int4*)(Alo + aoff);
        const int4 tHi = *(const int4*)(Ahi + aoff);
        const int2 tBi = *(const int2*)((const uint8_t*)Bi + boff);
        const int2 tBf = *(const int2*)((const uint8_t*)Bf + boff);
        const int2 tBg = *(const int2*)((const uint8_t*)Bg + boff);
        *(int4*)(dst + (jA << 10) + offA)       = tLo;
        *(int4*)(dst + ((4 + jA) << 10) + offA) = tHi;
        *(int2*)(dst + ((8 + nnB * 3 + 0) << 10) + offB) = tBi;
        *(int2*)(dst + ((8 + nnB * 3 + 1) << 10) + offB) = tBf;
        *(int2*)(dst + ((8 + nnB * 3 + 2) << 10) + offB) = tBg;
    };

    v8i accLoI = {}, accHiI = {};
    v8i accLoF = {}, accHiF = {};
    v8i accLoG = {}, accHiG = {};

    stage(sm[0], 0);
    __syncthreads();
    for (int kc = 0; kc < KC_; ++kc) {
        const uint8_t* cur = sm[kc & 1];
        if (kc + 1 < KC_) stage(sm[(kc + 1) & 1], kc + 1);
        const int lofs = lane * 32;
        const v8i alo = *(const v8i*)(cur + (mi << 10) + lofs);
        const v8i ahi = *(const v8i*)(cur + ((4 + mi) << 10) + lofs);
        const v8i bi  = *(const v8i*)(cur + ((8 + ni * 3 + 0) << 10) + lofs);
        const v8i bf  = *(const v8i*)(cur + ((8 + ni * 3 + 1) << 10) + lofs);
        const v8i bg  = *(const v8i*)(cur + ((8 + ni * 3 + 2) << 10) + lofs);
        accLoI = __builtin_amdgcn_wmma_i32_16x16x64_iu8(false, alo, true, bi, accLoI, false, false);
        accHiI = __builtin_amdgcn_wmma_i32_16x16x64_iu8(true,  ahi, true, bi, accHiI, false, false);
        accLoF = __builtin_amdgcn_wmma_i32_16x16x64_iu8(false, alo, true, bf, accLoF, false, false);
        accHiF = __builtin_amdgcn_wmma_i32_16x16x64_iu8(true,  ahi, true, bf, accHiF, false, false);
        accLoG = __builtin_amdgcn_wmma_i32_16x16x64_iu8(false, alo, true, bg, accLoG, false, false);
        accHiG = __builtin_amdgcn_wmma_i32_16x16x64_iu8(true,  ahi, true, bg, accHiG, false, false);
        __syncthreads();
    }

    const int mt = mtBase + mi;
    const int nt = ntBase + ni;
    const int col   = nt * 16 + (lane & 15);
    const int rbase = mt * 16 + ((lane & 16) >> 1);
    const int sci = si[col], scf = sf[col], scg = sg[col];
#pragma unroll
    for (int r = 0; r < 8; ++r) {
        const size_t o = (size_t)(rbase + r) * H_ + col;
        int ai = accLoI[r] + (accHiI[r] << 8);
        int af = accLoF[r] + (accHiF[r] << 8);
        int ag = accLoG[r] + (accHiG[r] << 8);
        int iv = clip16i((ai * sci) >> 8);
        int fv = clip16i((af * scf) >> 8);
        int gv = clip16i((ag * scg) >> 8);
        int fs = sigq(fv);
        int im = fmul16(fmul16(iv, sigq(iv)), (short)(256 - fs));
        FS[o] = (short)fs;
        IM[o] = (short)im;
        G[o]  = (short)gv;
    }
}

__global__ void __launch_bounds__(256)
scan_kernel(const float* __restrict__ hinit, short* __restrict__ FS,
            const short* __restrict__ IM, const short* __restrict__ G) {
    const int t = blockIdx.x * blockDim.x + threadIdx.x;
    const int b = t >> 10;
    float r = rintf(hinit[t] * 256.0f);
    r = fminf(fmaxf(r, -32768.0f), 32767.0f);
    int hv = (int)r;
    size_t idx = (size_t)b * S_ * H_ + (t & (H_ - 1));
#pragma unroll 8
    for (int s = 0; s < S_; ++s, idx += H_) {
        const int ft = FS[idx];
        const int it = IM[idx];
        const int gt = G[idx];
        hv = (short)(fmul16(ft, hv) + it);
        FS[idx] = (short)fmul16(gt, hv);
    }
}

__global__ void __launch_bounds__(256)
gemm_o_kernel(const uint8_t* __restrict__ Alo, const uint8_t* __restrict__ Ahi,
              const int8_t* __restrict__ Bo, const int* __restrict__ so,
              float* __restrict__ out) {
    __shared__ __align__(16) uint8_t sm[2][12 * 1024];
    const int tid  = threadIdx.x;
    const int lane = tid & 31;
    const int w    = tid >> 5;
    const int mtBase = (blockIdx.x >> 4) << 2;
    const int ntBase = (blockIdx.x & 15) << 2;
    const int ni  = w & 3;
    const int mi0 = (w >> 2) << 1;
    const int mi1 = mi0 + 1;

    const int jA   = tid >> 6;
    const int offA = (tid & 63) << 4;

    auto stage = [&](uint8_t* dst, int kc) {
        const size_t aoff = (((size_t)(mtBase + jA) * KC_ + kc) << 10) + offA;
        const size_t boff = (((size_t)(ntBase + jA) * KC_ + kc) << 10) + offA;
        const int4 tLo = *(const int4*)(Alo + aoff);
        const int4 tHi = *(const int4*)(Ahi + aoff);
        const int4 tBo = *(const int4*)((const uint8_t*)Bo + boff);
        *(int4*)(dst + (jA << 10) + offA)       = tLo;
        *(int4*)(dst + ((4 + jA) << 10) + offA) = tHi;
        *(int4*)(dst + ((8 + jA) << 10) + offA) = tBo;
    };

    v8i accLo0 = {}, accHi0 = {}, accLo1 = {}, accHi1 = {};

    stage(sm[0], 0);
    __syncthreads();
    for (int kc = 0; kc < KC_; ++kc) {
        const uint8_t* cur = sm[kc & 1];
        if (kc + 1 < KC_) stage(sm[(kc + 1) & 1], kc + 1);
        const int lofs = lane * 32;
        const v8i alo0 = *(const v8i*)(cur + (mi0 << 10) + lofs);
        const v8i ahi0 = *(const v8i*)(cur + ((4 + mi0) << 10) + lofs);
        const v8i alo1 = *(const v8i*)(cur + (mi1 << 10) + lofs);
        const v8i ahi1 = *(const v8i*)(cur + ((4 + mi1) << 10) + lofs);
        const v8i bo   = *(const v8i*)(cur + ((8 + ni) << 10) + lofs);
        accLo0 = __builtin_amdgcn_wmma_i32_16x16x64_iu8(false, alo0, true, bo, accLo0, false, false);
        accHi0 = __builtin_amdgcn_wmma_i32_16x16x64_iu8(true,  ahi0, true, bo, accHi0, false, false);
        accLo1 = __builtin_amdgcn_wmma_i32_16x16x64_iu8(false, alo1, true, bo, accLo1, false, false);
        accHi1 = __builtin_amdgcn_wmma_i32_16x16x64_iu8(true,  ahi1, true, bo, accHi1, false, false);
        __syncthreads();
    }

    const int nt  = ntBase + ni;
    const int col = nt * 16 + (lane & 15);
    const int sc  = so[col];
#pragma unroll
    for (int tl = 0; tl < 2; ++tl) {
        const int mt = mtBase + (tl ? mi1 : mi0);
        const int rbase = mt * 16 + ((lane & 16) >> 1);
#pragma unroll
        for (int r = 0; r < 8; ++r) {
            int lo = tl ? accLo1[r] : accLo0[r];
            int hi = tl ? accHi1[r] : accHi0[r];
            int acc = lo + (hi << 8);
            int ov  = clip16i((acc * sc) >> 8);
            out[(size_t)(rbase + r) * H_ + col] = (float)ov;
        }
    }
}

extern "C" void kernel_launch(void* const* d_in, const int* in_sizes, int n_in,
                              void* d_out, int out_size, void* d_ws, size_t ws_size,
                              hipStream_t stream) {
    const float* x    = (const float*)d_in[0];
    const int*   w_i  = (const int*)d_in[1];
    const int*   w_f  = (const int*)d_in[2];
    const int*   w_g  = (const int*)d_in[3];
    const int*   w_o  = (const int*)d_in[4];
    const int*   s_i  = (const int*)d_in[5];
    const int*   s_f  = (const int*)d_in[6];
    const int*   s_g  = (const int*)d_in[7];
    const int*   s_o  = (const int*)d_in[8];
    const float* hini = (const float*)d_in[9];
    float* out = (float*)d_out;

    uint8_t* ws = (uint8_t*)d_ws;
    int8_t*  pw_i = (int8_t*)(ws + WOFF_PW);
    int8_t*  pw_f = pw_i + (size_t)H_ * H_;
    int8_t*  pw_g = pw_f + (size_t)H_ * H_;
    int8_t*  pw_o = pw_g + (size_t)H_ * H_;
    uint8_t* alo  = ws + WOFF_ALO;
    uint8_t* ahi  = ws + WOFF_AHI;
    short*   FS   = (short*)(ws + WOFF_FS);
    short*   IM   = (short*)(ws + WOFF_IM);
    short*   G    = (short*)(ws + WOFF_G);

    dim3 pgrid(KC_, NT_);
    pack_w_kernel<<<pgrid, 256, 0, stream>>>(w_i, pw_i);
    pack_w_kernel<<<pgrid, 256, 0, stream>>>(w_f, pw_f);
    pack_w_kernel<<<pgrid, 256, 0, stream>>>(w_g, pw_g);
    pack_w_kernel<<<pgrid, 256, 0, stream>>>(w_o, pw_o);

    dim3 qgrid(KC_, MT_);
    quant_pack_x_kernel<<<qgrid, 256, 0, stream>>>(x, alo, ahi);

    gemm_ifg_kernel<<<(MT_ / 4) * (NT_ / 2), 256, 0, stream>>>(
        alo, ahi, pw_i, pw_f, pw_g, s_i, s_f, s_g, FS, IM, G);

    scan_kernel<<<(B_ * H_) / 256, 256, 0, stream>>>(hini, FS, IM, G);

    pack_gh_kernel<<<qgrid, 256, 0, stream>>>(FS, alo, ahi);

    gemm_o_kernel<<<(MT_ / 4) * (NT_ / 4), 256, 0, stream>>>(
        alo, ahi, pw_o, s_o, out);
}
